// MoETransformerBlock_89902255440749
// MI455X (gfx1250) — matrix operand order repaired and checked by reading, not compiled, not run on hardware
//
#include <hip/hip_runtime.h>

// ---------------------------------------------------------------------------
// MoE transformer block for MI455X (gfx1250, wave32, WMMA + TDM).
//   B=1, S=2048 (T=2048 tokens), H=1024, E=8 experts, top-K=2, F=4096.
// Pipeline:
//   k_zero      : zero expert counters
//   k_ln_gate   : layernorm -> bf16 tokens, gate logits, softmax, top-2 routing
//   k_route     : 64-aligned per-expert offsets, aux loss, row scatter
//   k_wconvert  : Wg/Wu/Wd f32 -> bf16 (one pass; GEMMs then stream bf16, L2-resident)
//   k_gemm1     : act = g * silu(u)   (64x128 tile, 8 WMMA/k-step/wave, reg pipeline)
//   k_gemm2     : y   = act @ Wd^T    (64x128 tile, 4 WMMA/k-step/wave, TDM A-staging)
//   k_combine   : out = x + w0*y[row0] + w1*y[row1]
// ---------------------------------------------------------------------------

typedef __attribute__((ext_vector_type(16))) __bf16 v16bf;
typedef __attribute__((ext_vector_type(4)))  __bf16 v4bf;
typedef __attribute__((ext_vector_type(8)))  float  v8f;

constexpr int kT = 2048;    // tokens
constexpr int kH = 1024;    // hidden
constexpr int kE = 8;       // experts
constexpr int kF = 4096;    // ffn dim
constexpr int kRows = 2 * kT;              // routed rows (token -> 2 experts)
constexpr int kMTile = 64;                 // rows per GEMM block (expert-aligned)
constexpr int kRowCap = kRows + kE * kMTile;   // 4608

#if __has_builtin(__builtin_amdgcn_tensor_load_to_lds)
#define USE_TDM 1
typedef unsigned int v4u __attribute__((ext_vector_type(4)));
typedef int          v8i __attribute__((ext_vector_type(8)));
typedef int          v4i __attribute__((ext_vector_type(4)));

// TDM load of a 2D tile: kMTile rows x 32 bf16, row stride = strideElems,
// into LDS at lds_addr (rows contiguous, 64B apart).  D# per ISA ch.8.
// clang-23 toolchain: 6-arg form (g0 v4u, g1 v8i, g2 v4i, g3 v4i, v8i, cpol).
__device__ inline void tdm_load_tile(const __bf16* gsrc, unsigned lds_addr,
                                     unsigned strideElems) {
    const unsigned long long ga = (unsigned long long)(uintptr_t)gsrc;   // bytes
    const unsigned td0 = 32u, td1 = (unsigned)kMTile;                    // tensor dims
    const unsigned tile0 = 32u, tile1 = (unsigned)kMTile;                // tile dims
    v4u g0;
    g0[0] = 1u;                                            // count=1, user descriptor
    g0[1] = lds_addr;                                      // lds_addr [63:32]
    g0[2] = (unsigned)ga;                                  // global_addr [95:64]
    g0[3] = (unsigned)((ga >> 32) & 0x1FFFFFFull)          // global_addr [120:96]
          | (2u << 30);                                    // type=2 ("image")
    v8i g1;
    g1[0] = (int)(1u << 16);                               // data_size=1 (2 bytes)
    g1[1] = (int)((td0 & 0xFFFFu) << 16);                  // tensor_dim0 -> bits[79:48]
    g1[2] = (int)((td0 >> 16) | ((td1 & 0xFFFFu) << 16));  // tensor_dim1 -> bits[111:80]
    g1[3] = (int)((td1 >> 16) | (tile0 << 16));            // tile_dim0   -> bits[127:112]
    g1[4] = (int)(tile1 & 0xFFFFu);                        // tile_dim1; tile_dim2=0
    g1[5] = (int)strideElems;                              // tensor_dim0_stride[31:0]
    g1[6] = 0;                                             // stride[47:32]=0; dim1_stride lo=0
    g1[7] = 0;
    const v4i z4 = {0, 0, 0, 0};                           // 2-D tensor: groups 2/3 unused
    const v8i z8 = {0, 0, 0, 0, 0, 0, 0, 0};
    __builtin_amdgcn_tensor_load_to_lds(g0, g1, z4, z4, z8, 0);
}
#else
#define USE_TDM 0
#endif

// ---------------------------------------------------------------------------
__global__ __launch_bounds__(64) void k_zero(unsigned int* counts) {
    if (threadIdx.x < kE) counts[threadIdx.x] = 0u;
}

// ---------------------------------------------------------------------------
// Weight pre-conversion f32 -> bf16 (grid-stride over float4).
__global__ __launch_bounds__(256) void k_wconvert(
    const float* __restrict__ src, __bf16* __restrict__ dst, long n4)
{
    long i = (long)blockIdx.x * 256 + threadIdx.x;
    const long stride = (long)gridDim.x * 256;
    for (; i < n4; i += stride) {
        const float4 q = ((const float4*)src)[i];
        v4bf o;
        o[0] = (__bf16)q.x; o[1] = (__bf16)q.y; o[2] = (__bf16)q.z; o[3] = (__bf16)q.w;
        ((v4bf*)dst)[i] = o;
    }
}

// ---------------------------------------------------------------------------
// One block per token: layernorm, bf16 cast, gate logits, softmax, top-2.
__global__ __launch_bounds__(256) void k_ln_gate(
    const float* __restrict__ x, const float* __restrict__ lnw,
    const float* __restrict__ lnb, const float* __restrict__ gw,
    __bf16* __restrict__ xnbf, unsigned int* __restrict__ counts,
    int* __restrict__ tok_e, int* __restrict__ tok_rank, float* __restrict__ tok_w)
{
    const int t = blockIdx.x;
    const int tid = threadIdx.x;
    __shared__ float red[256];
    __shared__ float logits_s[kE];

    const float* xr = x + (size_t)t * kH;
    float v[4];
    float s = 0.f, s2 = 0.f;
#pragma unroll
    for (int i = 0; i < 4; ++i) {
        v[i] = xr[tid + i * 256];
        s += v[i]; s2 += v[i] * v[i];
    }
    red[tid] = s; __syncthreads();
    for (int o = 128; o > 0; o >>= 1) { if (tid < o) red[tid] += red[tid + o]; __syncthreads(); }
    const float mean = red[0] * (1.0f / kH);
    __syncthreads();
    red[tid] = s2; __syncthreads();
    for (int o = 128; o > 0; o >>= 1) { if (tid < o) red[tid] += red[tid + o]; __syncthreads(); }
    const float var = red[0] * (1.0f / kH) - mean * mean;
    const float rstd = rsqrtf(var + 1e-5f);
    __syncthreads();

    float acc[kE];
#pragma unroll
    for (int e = 0; e < kE; ++e) acc[e] = 0.f;
#pragma unroll
    for (int i = 0; i < 4; ++i) {
        const int h = tid + i * 256;
        const float nv = (v[i] - mean) * rstd * lnw[h] + lnb[h];
        xnbf[(size_t)t * kH + h] = (__bf16)nv;
#pragma unroll
        for (int e = 0; e < kE; ++e) acc[e] += nv * gw[e * kH + h];
    }
#pragma unroll
    for (int e = 0; e < kE; ++e) {
        red[tid] = acc[e]; __syncthreads();
        for (int o = 128; o > 0; o >>= 1) { if (tid < o) red[tid] += red[tid + o]; __syncthreads(); }
        if (tid == 0) logits_s[e] = red[0];
        __syncthreads();
    }

    if (tid == 0) {
        float mx = logits_s[0];
        for (int e = 1; e < kE; ++e) mx = fmaxf(mx, logits_s[e]);
        float p[kE]; float sum = 0.f;
        for (int e = 0; e < kE; ++e) { p[e] = __expf(logits_s[e] - mx); sum += p[e]; }
        const float inv = 1.f / sum;
        int i0 = 0; float b0 = -1.f;
        for (int e = 0; e < kE; ++e) { p[e] *= inv; if (p[e] > b0) { b0 = p[e]; i0 = e; } }
        int i1 = 0; float b1 = -1.f;
        for (int e = 0; e < kE; ++e) { if (e != i0 && p[e] > b1) { b1 = p[e]; i1 = e; } }
        const float ssum = b0 + b1 + 1e-8f;
        const unsigned r0 = atomicAdd(&counts[i0], 1u);
        const unsigned r1 = atomicAdd(&counts[i1], 1u);
        tok_e[2 * t] = i0;        tok_e[2 * t + 1] = i1;
        tok_rank[2 * t] = (int)r0; tok_rank[2 * t + 1] = (int)r1;
        tok_w[2 * t] = b0 / ssum;  tok_w[2 * t + 1] = b1 / ssum;
    }
}

// ---------------------------------------------------------------------------
// Single block: 64-aligned prefix offsets, aux loss, row scatter.
__global__ __launch_bounds__(256) void k_route(
    const unsigned int* __restrict__ counts, unsigned int* __restrict__ offs,
    const int* __restrict__ tok_e, const int* __restrict__ tok_rank,
    int* __restrict__ tok_row, int* __restrict__ row_tok, int* __restrict__ row_expert,
    float* __restrict__ out_lbl)
{
    __shared__ unsigned int so[kE + 1];
    const int tid = threadIdx.x;
    if (tid == 0) {
        unsigned int a = 0;
        for (int e = 0; e < kE; ++e) { so[e] = a; a += (counts[e] + 63u) & ~63u; }
        so[kE] = a;
        for (int e = 0; e <= kE; ++e) offs[e] = so[e];
        // load-balance aux loss: (var_unbiased(usage) / (mean + eps))^2
        const float tot = (float)kRows;
        const float mu = 1.0f / kE;
        float var = 0.f;
        for (int e = 0; e < kE; ++e) { float d = counts[e] / tot - mu; var += d * d; }
        var /= (float)(kE - 1);
        const float l = var / (mu + 1e-8f);
        *out_lbl = l * l;
    }
    __syncthreads();
    for (int r = tid; r < kRowCap; r += 256) {
        int e = kE - 1;
        for (int q = 0; q < kE; ++q) {
            if ((unsigned)r >= so[q] && (unsigned)r < so[q + 1]) { e = q; break; }
        }
        row_expert[r] = e;
        row_tok[r] = 0;   // padded rows compute token 0 with gate weight 0
    }
    __syncthreads();
    for (int i = tid; i < kRows; i += 256) {
        const int e = tok_e[i];
        const int r = (int)so[e] + tok_rank[i];
        row_tok[r] = i >> 1;
        tok_row[i] = r;
    }
}

// ---------------------------------------------------------------------------
// Fragment helpers (wave32 WMMA; A and B both use contiguous k = 16*(lane/16) + s).
__device__ inline v16bf load_a_lds(const __bf16* __restrict__ rowp, int ak) {
    v16bf a;
#pragma unroll
    for (int i = 0; i < 8; ++i) { a[i] = rowp[ak + i]; a[8 + i] = rowp[ak + 8 + i]; }
    return a;
}
// 16 contiguous bf16 (32B, 32B-aligned): two global_load_b128, no cvt.
__device__ inline v16bf load_b_bf16(const __bf16* __restrict__ p) {
    return *(const v16bf*)p;
}
__device__ inline float fast_silu(float u) {
    return u * __builtin_amdgcn_rcpf(1.f + __expf(-u));
}

// ---------------------------------------------------------------------------
// GEMM1: act = g * silu(u).  Block tile: 64 rows x 128 f-cols, 8 waves.
// A rows are token-gathered (row_tok) -> register software pipeline.
__global__ __launch_bounds__(256) void k_gemm1(
    const __bf16* __restrict__ xnbf, const __bf16* __restrict__ Wg,
    const __bf16* __restrict__ Wu, const int* __restrict__ row_tok,
    const int* __restrict__ row_expert, const unsigned int* __restrict__ offs,
    __bf16* __restrict__ act)
{
    const int m0 = blockIdx.x * kMTile;
    if ((unsigned)m0 >= offs[kE]) return;
    const int e = row_expert[m0];
    const int tid = threadIdx.x;
    const int wave = tid >> 5, lane = tid & 31;
    const int n0 = blockIdx.y * 128 + wave * 16;

    __shared__ __align__(16) __bf16 As[2][kMTile * 32];

    v8f g[4], u[4];
#pragma unroll
    for (int s = 0; s < 4; ++s) { g[s] = (v8f){0,0,0,0,0,0,0,0}; u[s] = (v8f){0,0,0,0,0,0,0,0}; }

    const int bn = lane & 15;
    const int bk = (lane & 16) ? 16 : 0;   // B: lanes 0-15 K=0..15, lanes 16-31 K=16..31
    const int am = lane & 15;
    const int ak = (lane & 16) ? 16 : 0;   // A: K halves split at lane 16
    const int sr = tid >> 2;               // staged row 0..63 (one per thread)
    const int sc = (tid & 3) * 8;          // staged cols

    const __bf16* bgp = Wg + ((size_t)e * kF + (n0 + bn)) * kH + bk;
    const __bf16* bup = Wu + ((size_t)e * kF + (n0 + bn)) * kH + bk;
    const __bf16* xsrc = xnbf + (size_t)row_tok[m0 + sr] * kH + sc;
    __bf16* const sdst0 = &As[0][sr * 32 + sc];
    __bf16* const sdst1 = &As[1][sr * 32 + sc];

    uint4 stage = *(const uint4*)(xsrc);           // k0 = 0 tile
    *(uint4*)sdst0 = stage;

    auto gstep = [&](const __bf16* bufC, __bf16* sdstN, int k0) {
        const bool more = (k0 + 32) < kH;
        if (more) stage = *(const uint4*)(xsrc + k0 + 32);   // issue early, no wait
        __syncthreads();                                     // bufC stores visible
        const v16bf bg = load_b_bf16(bgp + k0);
        const v16bf bu = load_b_bf16(bup + k0);
        v16bf a[4];
#pragma unroll
        for (int s = 0; s < 4; ++s) a[s] = load_a_lds(&bufC[(s * 16 + am) * 32], ak);
#pragma unroll
        for (int s = 0; s < 4; ++s) {
            g[s] = __builtin_amdgcn_wmma_f32_16x16x32_bf16(false, a[s], false, bg, (short)0, g[s], false, false);
            u[s] = __builtin_amdgcn_wmma_f32_16x16x32_bf16(false, a[s], false, bu, (short)0, u[s], false, false);
        }
        if (more) *(uint4*)sdstN = stage;                    // waits loadcnt here, after WMMA issue
    };

#pragma unroll 1
    for (int k0 = 0; k0 < kH; k0 += 64) {
        gstep(As[0], sdst1, k0);
        gstep(As[1], sdst0, k0 + 32);
    }

    // epilogue: act = g * silu(u).  C/D layout: VGPR i -> M=i (+8 for hi lanes)
    const int crow = (lane & 16) ? 8 : 0;
    const int ccol = n0 + (lane & 15);
#pragma unroll
    for (int s = 0; s < 4; ++s) {
#pragma unroll
        for (int i = 0; i < 8; ++i) {
            const float a = g[s][i] * fast_silu(u[s][i]);
            act[(size_t)(m0 + s * 16 + crow + i) * kF + ccol] = (__bf16)a;
        }
    }
}

// ---------------------------------------------------------------------------
// GEMM2: y = act @ Wd[e]^T.  Block tile: 64 rows x 128 h-cols, K over F.
// A rows are contiguous -> Tensor Data Mover stages them async into LDS.
__global__ __launch_bounds__(256) void k_gemm2(
    const __bf16* __restrict__ act, const __bf16* __restrict__ Wd,
    const int* __restrict__ row_expert, const unsigned int* __restrict__ offs,
    float* __restrict__ y)
{
    const int m0 = blockIdx.x * kMTile;
    if ((unsigned)m0 >= offs[kE]) return;
    const int e = row_expert[m0];
    const int tid = threadIdx.x;
    const int wave = tid >> 5, lane = tid & 31;
    const int n0 = blockIdx.y * 128 + wave * 16;   // h columns

    __shared__ __align__(16) __bf16 As[2][kMTile * 32];

    v8f yv[4];
#pragma unroll
    for (int s = 0; s < 4; ++s) yv[s] = (v8f){0,0,0,0,0,0,0,0};

    const int bn = lane & 15;
    const int bk = (lane & 16) ? 16 : 0;
    const int am = lane & 15;
    const int ak = (lane & 16) ? 16 : 0;

    const __bf16* bdp = Wd + ((size_t)e * kH + (n0 + bn)) * kF + bk;
    const __bf16* atile = act + (size_t)m0 * kF;   // 64 contiguous rows, stride kF

#if USE_TDM
    const unsigned lds0 = (unsigned)(uintptr_t)&As[0][0];
    const unsigned lds1 = (unsigned)(uintptr_t)&As[1][0];
    if (wave == 0) tdm_load_tile(atile, lds0, (unsigned)kF);   // prologue: k0 = 0

    auto gstep = [&](const __bf16* bufC, unsigned ldsN, int k0) {
        if (wave == 0) __builtin_amdgcn_s_wait_tensorcnt(0);   // bufC TDM complete
        __syncthreads();                                       // prev reads of next buf drained
        if (wave == 0 && (k0 + 32) < kF)
            tdm_load_tile(atile + k0 + 32, ldsN, (unsigned)kF);
        const v16bf b = load_b_bf16(bdp + k0);
        v16bf a[4];
#pragma unroll
        for (int s = 0; s < 4; ++s) a[s] = load_a_lds(&bufC[(s * 16 + am) * 32], ak);
#pragma unroll
        for (int s = 0; s < 4; ++s)
            yv[s] = __builtin_amdgcn_wmma_f32_16x16x32_bf16(false, a[s], false, b, (short)0, yv[s], false, false);
    };
#else
    const int sr = tid >> 2;
    const int sc = (tid & 3) * 8;
    const __bf16* asrc = atile + (size_t)sr * kF + sc;
    __bf16* const sdst0 = &As[0][sr * 32 + sc];
    __bf16* const sdst1 = &As[1][sr * 32 + sc];
    uint4 stage = *(const uint4*)(asrc);
    *(uint4*)sdst0 = stage;

    auto gstep = [&](const __bf16* bufC, __bf16* sdstN, int k0) {
        const bool more = (k0 + 32) < kF;
        if (more) stage = *(const uint4*)(asrc + k0 + 32);
        __syncthreads();
        const v16bf b = load_b_bf16(bdp + k0);
        v16bf a[4];
#pragma unroll
        for (int s = 0; s < 4; ++s) a[s] = load_a_lds(&bufC[(s * 16 + am) * 32], ak);
#pragma unroll
        for (int s = 0; s < 4; ++s)
            yv[s] = __builtin_amdgcn_wmma_f32_16x16x32_bf16(false, a[s], false, b, (short)0, yv[s], false, false);
        if (more) *(uint4*)sdstN = stage;
    };
#endif

#pragma unroll 1
    for (int k0 = 0; k0 < kF; k0 += 64) {
#if USE_TDM
        gstep(As[0], lds1, k0);
        gstep(As[1], lds0, k0 + 32);
#else
        gstep(As[0], sdst1, k0);
        gstep(As[1], sdst0, k0 + 32);
#endif
    }

    const int crow = (lane & 16) ? 8 : 0;
    const int ccol = n0 + (lane & 15);
#pragma unroll
    for (int s = 0; s < 4; ++s)
#pragma unroll
        for (int i = 0; i < 8; ++i)
            y[(size_t)(m0 + s * 16 + crow + i) * kH + ccol] = yv[s][i];
}

// ---------------------------------------------------------------------------
// Combine: out = x + w0*y[row0] + w1*y[row1]  (deterministic, no atomics).
__global__ __launch_bounds__(256) void k_combine(
    const float* __restrict__ x, const float* __restrict__ yb,
    const int* __restrict__ tok_row, const float* __restrict__ tok_w,
    float* __restrict__ out)
{
    const int t = blockIdx.x;
    const int r0 = tok_row[2 * t], r1 = tok_row[2 * t + 1];
    const float w0 = tok_w[2 * t], w1 = tok_w[2 * t + 1];
    const int h = threadIdx.x * 4;
    const float4 xv = *(const float4*)(x  + (size_t)t  * kH + h);
    const float4 a  = *(const float4*)(yb + (size_t)r0 * kH + h);
    const float4 b  = *(const float4*)(yb + (size_t)r1 * kH + h);
    float4 o;
    o.x = xv.x + w0 * a.x + w1 * b.x;
    o.y = xv.y + w0 * a.y + w1 * b.y;
    o.z = xv.z + w0 * a.z + w1 * b.z;
    o.w = xv.w + w0 * a.w + w1 * b.w;
    *(float4*)(out + (size_t)t * kH + h) = o;
}

// ---------------------------------------------------------------------------
extern "C" void kernel_launch(void* const* d_in, const int* in_sizes, int n_in,
                              void* d_out, int out_size, void* d_ws, size_t ws_size,
                              hipStream_t stream) {
    const float* x   = (const float*)d_in[0];
    const float* lnw = (const float*)d_in[1];
    const float* lnb = (const float*)d_in[2];
    const float* gw  = (const float*)d_in[3];
    const float* Wg  = (const float*)d_in[4];
    const float* Wu  = (const float*)d_in[5];
    const float* Wd  = (const float*)d_in[6];
    float* out = (float*)d_out;

    char* ws = (char*)d_ws;
    size_t off = 0;
    auto take = [&](size_t bytes) -> char* {
        char* p = ws + off;
        off = (off + bytes + 255) & ~(size_t)255;
        return p;
    };
    const size_t wElems = (size_t)kE * kF * kH;        // per weight tensor
    __bf16*       xnbf     = (__bf16*)       take((size_t)kT * kH * 2);
    unsigned int* counts   = (unsigned int*) take(64);
    unsigned int* offs     = (unsigned int*) take(64);
    int*          tok_e    = (int*)          take((size_t)kRows * 4);
    int*          tok_rank = (int*)          take((size_t)kRows * 4);
    float*        tok_w    = (float*)        take((size_t)kRows * 4);
    int*          tok_row  = (int*)          take((size_t)kRows * 4);
    int*          row_tok  = (int*)          take((size_t)kRowCap * 4);
    int*          row_exp  = (int*)          take((size_t)kRowCap * 4);
    __bf16*       wg_bf    = (__bf16*)       take(wElems * 2);
    __bf16*       wu_bf    = (__bf16*)       take(wElems * 2);
    __bf16*       wd_bf    = (__bf16*)       take(wElems * 2);
    __bf16*       act      = (__bf16*)       take((size_t)kRowCap * kF * 2);
    float*        yb       = (float*)        take((size_t)kRowCap * kH * 4);
    (void)in_sizes; (void)n_in; (void)out_size; (void)ws_size;

    const long n4 = (long)(wElems / 4);
    k_zero<<<1, 64, 0, stream>>>(counts);
    k_wconvert<<<2048, 256, 0, stream>>>(Wg, wg_bf, n4);
    k_wconvert<<<2048, 256, 0, stream>>>(Wu, wu_bf, n4);
    k_wconvert<<<2048, 256, 0, stream>>>(Wd, wd_bf, n4);
    k_ln_gate<<<kT, 256, 0, stream>>>(x, lnw, lnb, gw, xnbf, counts, tok_e, tok_rank, tok_w);
    k_route<<<1, 256, 0, stream>>>(counts, offs, tok_e, tok_rank,
                                   tok_row, row_tok, row_exp, out + (size_t)kT * kH);
    k_gemm1<<<dim3(kRowCap / kMTile, kF / 128), 256, 0, stream>>>(
        xnbf, wg_bf, wu_bf, row_tok, row_exp, offs, act);
    k_gemm2<<<dim3(kRowCap / kMTile, kH / 128), 256, 0, stream>>>(
        act, wd_bf, row_exp, offs, yb);
    k_combine<<<kT, 256, 0, stream>>>(x, yb, tok_row, tok_w, out);
}
